// Seq2seqEnsemble_27659589386605
// MI455X (gfx1250) — compile-verified, matrix operand order checked by reading, not run on hardware
//
#include <hip/hip_runtime.h>
#include <math.h>
#include <stdint.h>

// ---------------- problem constants (from reference) ----------------
#define Bq   32
#define Tq   32
#define Eq   1024
#define Hq   1024
#define G4H  4096
#define Vq   32000
#define Lq   2
#define SEQ  20
#define TOPN 10
#define BH   (Bq * Hq)

// ---------------- WMMA types ----------------
typedef __attribute__((ext_vector_type(16))) _Float16 v16h;
typedef __attribute__((ext_vector_type(8)))  float    v8f;

// 16-bit A/B fragment K-index map (per CDNA5 ISA 16-bit 16x32 layout):
// element e of lane L -> K = (e/8)*16 + (L/16)*8 + (e%8)
__device__ __forceinline__ int kmap(int e, int lane) {
    return ((e >> 3) << 4) + ((lane >> 4) << 3) + (e & 7);
}

// A fragment from LDS-staged f16 tile (row stride 40 halfs = 80B, 16B aligned ->
// the two 8-contiguous-half chunks merge into ds_load_b128)
__device__ __forceinline__ v16h load_a_lds(const _Float16 (*lds)[40], int mt, int lane) {
    v16h a;
#pragma unroll
    for (int e = 0; e < 16; ++e)
        a[e] = lds[mt * 16 + (lane & 15)][kmap(e, lane)];
    return a;
}

// B fragment from pre-swizzled f16 weights: layout [N/16][K/32][lane][16 halfs].
// Each lane reads its own contiguous 32B -> two coalesced global_load_b128.
__device__ __forceinline__ v16h load_b_packed(const _Float16* __restrict__ Wsw,
                                              int ksteps, int n, int k0, int lane) {
    const size_t blk = ((size_t)(n >> 4) * ksteps + (k0 >> 5)) * 512 + (size_t)lane * 16;
    return *(const v16h*)(Wsw + blk);
}

__device__ __forceinline__ float sigf(float x) { return 1.0f / (1.0f + expf(-x)); }

// ---------------- one-shot weight pack: f32 [K,N] -> f16 fragment-swizzled ----------------
// grid: (N/16, K/32), block 256. Output writes fully coalesced.
__global__ __launch_bounds__(256) void pack_b_kernel(const float* __restrict__ W,
                                                     _Float16* __restrict__ Wsw,
                                                     int N, int K) {
    const int ntile = blockIdx.x;
    const int ks    = blockIdx.y;
    _Float16* dst = Wsw + ((size_t)ntile * (K >> 5) + ks) * 512;
#pragma unroll
    for (int i = threadIdx.x; i < 512; i += 256) {
        const int lane = i >> 4, e = i & 15;
        const int n = ntile * 16 + (lane & 15);
        const int k = ks * 32 + kmap(e, lane);
        dst[i] = (_Float16)W[(size_t)k * N + n];
    }
}

// ---------------- generic WMMA GEMM: C[M,N] = A[M,K] * Wsw (+bias) ----------------
// grid: (N/256, M/32), block 256 (8 waves). Wave w: m-tile = w&1, 64-col strip = (w>>1)*64.
__global__ __launch_bounds__(256) void wmma_gemm_kernel(
    const float* __restrict__ A, int lda,
    const _Float16* __restrict__ Wsw,      // packed [N/16][K/32][32][16]
    const float* __restrict__ bias,
    float* __restrict__ C, long long ldc,
    int K)
{
    __shared__ _Float16 lds_a[32][40];
    const int tid  = threadIdx.x;
    const int lane = tid & 31;
    const int w    = tid >> 5;
    const int mt   = w & 1;
    const int wn   = (w >> 1) * 64;
    const int m0   = blockIdx.y * 32;
    const int nblk = blockIdx.x * 256;
    const int ksteps = K >> 5;

    v8f acc[4];
#pragma unroll
    for (int nt = 0; nt < 4; ++nt)
#pragma unroll
        for (int e = 0; e < 8; ++e) acc[nt][e] = 0.0f;

    for (int k0 = 0; k0 < K; k0 += 32) {
        // stage A (32x32 f32 -> f16), one float4 per thread
        {
            const int r = tid >> 3, kk = (tid & 7) * 4;
            const float4 f = *(const float4*)(A + (size_t)(m0 + r) * lda + k0 + kk);
            lds_a[r][kk + 0] = (_Float16)f.x;
            lds_a[r][kk + 1] = (_Float16)f.y;
            lds_a[r][kk + 2] = (_Float16)f.z;
            lds_a[r][kk + 3] = (_Float16)f.w;
        }
        if (k0 + 32 < K)
            __builtin_prefetch(Wsw + ((size_t)((nblk + wn) >> 4) * ksteps + ((k0 >> 5) + 1)) * 512
                               + (size_t)lane * 16, 0, 1);
        __syncthreads();

        const v16h a = load_a_lds(lds_a, mt, lane);
#pragma unroll
        for (int nt = 0; nt < 4; ++nt) {
            const v16h b = load_b_packed(Wsw, ksteps, nblk + wn + nt * 16, k0, lane);
            acc[nt] = __builtin_amdgcn_wmma_f32_16x16x32_f16(
                false, a, false, b, (short)0, acc[nt], false, false);
        }
        __syncthreads();
    }

#pragma unroll
    for (int nt = 0; nt < 4; ++nt) {
        const int col = nblk + wn + nt * 16 + (lane & 15);
        const float bv = bias ? bias[col] : 0.0f;
#pragma unroll
        for (int r = 0; r < 8; ++r) {
            const int row = m0 + mt * 16 + r + ((lane >> 4) << 3);
            C[(long long)row * ldc + col] = acc[nt][r] + bv;
        }
    }
}

// ---------------- fused LSTM cell step ----------------
// z = h_in@Wh (+ x_in@Wx | + xwx_t) + b ; gates -> (h_out, c_out), optional seq_out.
// grid: 16 blocks x 256 threads; block owns 64 gate-columns replicated across 4 gates.
// Wave w: m-tile = w&1, gate = w>>1 (i/f/g/o for matching cols meet in LDS).
__global__ __launch_bounds__(256) void lstm_step_kernel(
    const float* __restrict__ h_in, const float* __restrict__ c_in,
    const _Float16* __restrict__ WhP,                // packed [4H/16][H/32][32][16]
    const float* __restrict__ x_in,                  // [B, E] or nullptr
    const _Float16* __restrict__ WxP,                // packed [4H/16][E/32][32][16]
    const float* __restrict__ xwx, int xwx_ld,       // precomputed x@Wx slice or nullptr
    const float* __restrict__ bias,                  // [4H]
    float* __restrict__ h_out, float* __restrict__ c_out,
    float* __restrict__ seq_out, long long seq_ld)
{
    __shared__ _Float16 lds_a[32][40];
    __shared__ float zsm[4][32][64];
    const int tid  = threadIdx.x;
    const int lane = tid & 31;
    const int w    = tid >> 5;
    const int mt   = w & 1;
    const int gate = w >> 1;
    const int j0   = blockIdx.x * 64;
    const int ksteps = Hq >> 5;

    v8f acc[4];
#pragma unroll
    for (int nt = 0; nt < 4; ++nt)
#pragma unroll
        for (int e = 0; e < 8; ++e) acc[nt][e] = 0.0f;

#pragma unroll 1
    for (int pass = 0; pass < 2; ++pass) {
        const float*    Ap = (pass == 0) ? h_in : x_in;
        const _Float16* Wp = (pass == 0) ? WhP : WxP;
        if (!Ap) continue;
        for (int k0 = 0; k0 < Hq; k0 += 32) {
            // stage A (32x32)
            {
                const int r = tid >> 3, kk = (tid & 7) * 4;
                const float4 f = *(const float4*)(Ap + (size_t)r * Hq + k0 + kk);
                lds_a[r][kk + 0] = (_Float16)f.x;
                lds_a[r][kk + 1] = (_Float16)f.y;
                lds_a[r][kk + 2] = (_Float16)f.z;
                lds_a[r][kk + 3] = (_Float16)f.w;
            }
            if (k0 + 32 < Hq)
                __builtin_prefetch(Wp + ((size_t)((gate * Hq + j0) >> 4) * ksteps + ((k0 >> 5) + 1)) * 512
                                   + (size_t)lane * 16, 0, 1);
            __syncthreads();

            const v16h a = load_a_lds(lds_a, mt, lane);
#pragma unroll
            for (int nt = 0; nt < 4; ++nt) {
                const v16h b = load_b_packed(Wp, ksteps, gate * Hq + j0 + nt * 16, k0, lane);
                acc[nt] = __builtin_amdgcn_wmma_f32_16x16x32_f16(
                    false, a, false, b, (short)0, acc[nt], false, false);
            }
            __syncthreads();
        }
    }

    // ---- park z (+bias +xwx) in LDS for the gate exchange ----
#pragma unroll
    for (int nt = 0; nt < 4; ++nt) {
        const int lc   = nt * 16 + (lane & 15);
        const int zcol = gate * Hq + j0 + lc;
        const float bv = bias[zcol];
#pragma unroll
        for (int r = 0; r < 8; ++r) {
            const int row = mt * 16 + r + ((lane >> 4) << 3);
            float v = acc[nt][r] + bv;
            if (xwx) v += xwx[(size_t)row * xwx_ld + zcol];
            zsm[gate][row][lc] = v;
        }
    }
    __syncthreads();

    // ---- pointwise LSTM update: 32 rows x 64 cols, 8 per thread ----
#pragma unroll
    for (int e = 0; e < 8; ++e) {
        const int idx  = tid * 8 + e;
        const int row  = idx >> 6;
        const int col  = idx & 63;
        const int hcol = j0 + col;
        const float iv = zsm[0][row][col];
        const float fv = zsm[1][row][col];
        const float gv = zsm[2][row][col];
        const float ov = zsm[3][row][col];
        const float cp = c_in[row * Hq + hcol];
        const float cn = sigf(fv) * cp + sigf(iv) * tanhf(gv);
        const float hn = sigf(ov) * tanhf(cn);
        c_out[row * Hq + hcol] = cn;
        h_out[row * Hq + hcol] = hn;
        if (seq_out) seq_out[(long long)row * seq_ld + hcol] = hn;
    }
}

// ---------------- misc kernels ----------------
__global__ void zero_kernel(float* p, int n) {
    const int i = blockIdx.x * blockDim.x + threadIdx.x;
    if (i < n) p[i] = 0.0f;
}

__global__ void embed_seq_kernel(const int* __restrict__ tokens,
                                 const float* __restrict__ emb1,
                                 const float* __restrict__ emb2,
                                 float* __restrict__ x1, float* __restrict__ x2) {
    const int bt  = blockIdx.x;               // b*T + t
    const int tok = tokens[bt];
    const float* s1 = emb1 + (size_t)tok * Eq;
    const float* s2 = emb2 + (size_t)tok * Eq;
    float* d1 = x1 + (size_t)bt * Eq;
    float* d2 = x2 + (size_t)bt * Eq;
    for (int e = threadIdx.x; e < Eq; e += blockDim.x) { d1[e] = s1[e]; d2[e] = s2[e]; }
}

__global__ void dec_embed_kernel(const int* __restrict__ tok, int const_tok,
                                 const float* __restrict__ emb, float* __restrict__ x) {
    const int b = blockIdx.x;
    const int t = tok ? tok[b] : const_tok;
    const float* s = emb + (size_t)t * Eq;
    float* d = x + (size_t)b * Eq;
    for (int e = threadIdx.x; e < Eq; e += blockDim.x) d[e] = s[e];
}

// out = (s1 + 2*s2) / (||s1 + 2*s2||_2 + eps)   over all B*H elements
__global__ __launch_bounds__(1024) void blend_norm_kernel(
    const float* __restrict__ s1, const float* __restrict__ s2, float* __restrict__ out) {
    __shared__ float red[32];
    const int tid = threadIdx.x;
    float ss = 0.0f;
    for (int i = tid; i < BH; i += 1024) {
        const float v = s1[i] + 2.0f * s2[i];
        ss += v * v;
    }
#pragma unroll
    for (int off = 16; off; off >>= 1) ss += __shfl_xor(ss, off, 32);
    if ((tid & 31) == 0) red[tid >> 5] = ss;
    __syncthreads();
    if (tid < 32) {
        float v = red[tid];
#pragma unroll
        for (int off = 16; off; off >>= 1) v += __shfl_xor(v, off, 32);
        if (tid == 0) red[0] = sqrtf(v);
    }
    __syncthreads();
    const float inv = 1.0f / (red[0] + 1e-12f);
    for (int i = tid; i < BH; i += 1024)
        out[i] = (s1[i] + 2.0f * s2[i]) * inv;
}

__global__ __launch_bounds__(1024) void argmax_kernel(
    const float* __restrict__ logits, long long ldrow,
    int* __restrict__ tok, float* __restrict__ tokf, int step) {
    const int b = threadIdx.x >> 5, lane = threadIdx.x & 31;
    const float* row = logits + (long long)b * ldrow;
    float best = -3.4e38f; int bidx = 0x7fffffff;
    for (int j = lane; j < Vq; j += 32) {
        const float v = row[j];
        if (v > best || (v == best && j < bidx)) { best = v; bidx = j; }
    }
#pragma unroll
    for (int off = 16; off; off >>= 1) {
        const float ov = __shfl_xor(best, off, 32);
        const int   oi = __shfl_xor(bidx, off, 32);
        if (ov > best || (ov == best && oi < bidx)) { best = ov; bidx = oi; }
    }
    if (lane == 0) { tok[b] = bidx; tokf[b * SEQ + step] = (float)bidx; }
}

// top-n (n=10) by iterated masked wave-argmax, then normalized-raw-logit sampling
__global__ __launch_bounds__(1024) void topk_sample_kernel(
    const float* __restrict__ logits, long long ldrow,
    int* __restrict__ tok, float* __restrict__ tokf, int step) {
    const int b = threadIdx.x >> 5, lane = threadIdx.x & 31;
    const float* row = logits + (long long)b * ldrow;
    float vals[TOPN]; int idxs[TOPN];
#pragma unroll
    for (int it = 0; it < TOPN; ++it) {
        float best = -3.4e38f; int bidx = 0x7fffffff;
        for (int j = lane; j < Vq; j += 32) {
            bool skip = false;
#pragma unroll
            for (int k = 0; k < TOPN; ++k)
                if (k < it && idxs[k] == j) skip = true;
            const float v = row[j];
            if (!skip && (v > best || (v == best && j < bidx))) { best = v; bidx = j; }
        }
#pragma unroll
        for (int off = 16; off; off >>= 1) {
            const float ov = __shfl_xor(best, off, 32);
            const int   oi = __shfl_xor(bidx, off, 32);
            if (ov > best || (ov == best && oi < bidx)) { best = ov; bidx = oi; }
        }
        vals[it] = best; idxs[it] = bidx;
    }
    if (lane == 0) {
        float s = 0.0f;
#pragma unroll
        for (int k = 0; k < TOPN; ++k) s += vals[k];
        const float inv = 1.0f / s;
        unsigned h = (unsigned)(step * 0x9E3779B9u) ^ (unsigned)(b * 0x85EBCA6Bu) ^ 0xC2B2AE35u;
        h ^= h >> 16; h *= 0x7FEB352Du; h ^= h >> 15; h *= 0x846CA68Bu; h ^= h >> 16;
        const float u = (float)(h & 0xFFFFFFu) * (1.0f / 16777216.0f);
        float csum = 0.0f; int j = 0;
#pragma unroll
        for (int k = 0; k < TOPN; ++k) { csum += vals[k] * inv; if (csum < u) ++j; }
        if (j > TOPN - 1) j = TOPN - 1;
        const int t = idxs[j];
        tok[b] = t; tokf[b * SEQ + step] = (float)t;
    }
}

// ---------------- host orchestration ----------------
extern "C" void kernel_launch(void* const* d_in, const int* in_sizes, int n_in,
                              void* d_out, int out_size, void* d_ws, size_t ws_size,
                              hipStream_t stream) {
    (void)in_sizes; (void)n_in; (void)out_size; (void)ws_size;
    const int*   tokens = (const int*)d_in[0];
    const float* emb1   = (const float*)d_in[1];
    const float* emb2   = (const float*)d_in[2];
    const float* e1Wx   = (const float*)d_in[3];
    const float* e1Wh   = (const float*)d_in[4];
    const float* e1b    = (const float*)d_in[5];
    const float* e2Wx   = (const float*)d_in[6];
    const float* e2Wh   = (const float*)d_in[7];
    const float* e2b    = (const float*)d_in[8];
    const float* dWx    = (const float*)d_in[9];
    const float* dWh    = (const float*)d_in[10];
    const float* dbias  = (const float*)d_in[11];
    const float* Wd     = (const float*)d_in[12];
    const float* bd     = (const float*)d_in[13];
    float* out = (float*)d_out;

    // ---- workspace carving: f32 region ----
    float* ws = (float*)d_ws;
    float* x1a = ws; ws += (size_t)Bq * Tq * Eq;
    float* x1b = ws; ws += (size_t)Bq * Tq * Eq;
    float* x2a = ws; ws += (size_t)Bq * Tq * Eq;
    float* x2b = ws; ws += (size_t)Bq * Tq * Eq;
    float* xwx = ws; ws += (size_t)Bq * Tq * G4H;   // reused by enc1 then enc2 per layer
    float* h1A = ws; ws += BH;  float* h1B = ws; ws += BH;
    float* c1A = ws; ws += BH;  float* c1B = ws; ws += BH;
    float* h2A = ws; ws += BH;  float* h2B = ws; ws += BH;
    float* c2A = ws; ws += BH;  float* c2B = ws; ws += BH;
    float* HsA = ws; ws += (size_t)Lq * BH;  float* HsB = ws; ws += (size_t)Lq * BH;
    float* CsA = ws; ws += (size_t)Lq * BH;  float* CsB = ws; ws += (size_t)Lq * BH;
    float* xdec = ws; ws += (size_t)Bq * Eq;
    int* tokbuf = (int*)ws;

    // ---- f16 region: fragment-swizzled weights (64B aligned) ----
    char* hp = (char*)(tokbuf + 64);
    hp = (char*)(((uintptr_t)hp + 63) & ~(uintptr_t)63);
    const size_t WLSZ = (size_t)Eq * G4H;            // per-layer element count
    _Float16* e1WxP = (_Float16*)hp; hp += Lq * WLSZ * sizeof(_Float16);
    _Float16* e1WhP = (_Float16*)hp; hp += Lq * WLSZ * sizeof(_Float16);
    _Float16* e2WxP = (_Float16*)hp; hp += Lq * WLSZ * sizeof(_Float16);
    _Float16* e2WhP = (_Float16*)hp; hp += Lq * WLSZ * sizeof(_Float16);
    _Float16* dWxP  = (_Float16*)hp; hp += Lq * WLSZ * sizeof(_Float16);
    _Float16* dWhP  = (_Float16*)hp; hp += Lq * WLSZ * sizeof(_Float16);
    _Float16* WdP   = (_Float16*)hp;

    const dim3 blk256(256);
    const dim3 packGrid4(G4H / 16, Eq / 32);             // (256, 32)
    const dim3 packGridV(Vq / 16, Hq / 32);              // (2000, 32)
    const dim3 encGemmGrid(G4H / 256, (Bq * Tq) / 32);   // (16, 32)
    const dim3 lstmGrid(Hq / 64);                        // 16
    const dim3 logitsGrid(Vq / 256, 1);                  // (125, 1)
    const dim3 zeroGrid((BH + 255) / 256);

    // ---- one-shot weight packing (f32 row-major -> f16 fragment-swizzled) ----
    for (int l = 0; l < Lq; ++l) {
        pack_b_kernel<<<packGrid4, blk256, 0, stream>>>(e1Wx + l * WLSZ, e1WxP + l * WLSZ, G4H, Eq);
        pack_b_kernel<<<packGrid4, blk256, 0, stream>>>(e1Wh + l * WLSZ, e1WhP + l * WLSZ, G4H, Hq);
        pack_b_kernel<<<packGrid4, blk256, 0, stream>>>(e2Wx + l * WLSZ, e2WxP + l * WLSZ, G4H, Eq);
        pack_b_kernel<<<packGrid4, blk256, 0, stream>>>(e2Wh + l * WLSZ, e2WhP + l * WLSZ, G4H, Hq);
        pack_b_kernel<<<packGrid4, blk256, 0, stream>>>(dWx + l * WLSZ, dWxP + l * WLSZ, G4H, Eq);
        pack_b_kernel<<<packGrid4, blk256, 0, stream>>>(dWh + l * WLSZ, dWhP + l * WLSZ, G4H, Hq);
    }
    pack_b_kernel<<<packGridV, blk256, 0, stream>>>(Wd, WdP, Vq, Hq);

    // ---- encoders ----
    embed_seq_kernel<<<Bq * Tq, blk256, 0, stream>>>(tokens, emb1, emb2, x1a, x2a);

    float *xin1 = x1a, *xout1 = x1b, *xin2 = x2a, *xout2 = x2b;
    for (int l = 0; l < Lq; ++l) {
        // -------- encoder 1, layer l --------
        wmma_gemm_kernel<<<encGemmGrid, blk256, 0, stream>>>(
            xin1, Eq, e1WxP + l * WLSZ, nullptr, xwx, G4H, Eq);
        zero_kernel<<<zeroGrid, blk256, 0, stream>>>(h1A, BH);
        zero_kernel<<<zeroGrid, blk256, 0, stream>>>(c1A, BH);
        float *hi = h1A, *ho = h1B, *ci = c1A, *co = c1B;
        for (int t = 0; t < Tq; ++t) {
            lstm_step_kernel<<<lstmGrid, blk256, 0, stream>>>(
                hi, ci, e1WhP + l * WLSZ,
                nullptr, nullptr,
                xwx + (size_t)t * G4H, Tq * G4H,
                e1b + (size_t)l * G4H,
                ho, co, xout1 + (size_t)t * Eq, (long long)Tq * Eq);
            float* tmp;
            tmp = hi; hi = ho; ho = tmp;
            tmp = ci; ci = co; co = tmp;
        }
        float *h1fin = hi, *c1fin = ci;

        // -------- encoder 2, layer l --------
        wmma_gemm_kernel<<<encGemmGrid, blk256, 0, stream>>>(
            xin2, Eq, e2WxP + l * WLSZ, nullptr, xwx, G4H, Eq);
        zero_kernel<<<zeroGrid, blk256, 0, stream>>>(h2A, BH);
        zero_kernel<<<zeroGrid, blk256, 0, stream>>>(c2A, BH);
        hi = h2A; ho = h2B; ci = c2A; co = c2B;
        for (int t = 0; t < Tq; ++t) {
            lstm_step_kernel<<<lstmGrid, blk256, 0, stream>>>(
                hi, ci, e2WhP + l * WLSZ,
                nullptr, nullptr,
                xwx + (size_t)t * G4H, Tq * G4H,
                e2b + (size_t)l * G4H,
                ho, co, xout2 + (size_t)t * Eq, (long long)Tq * Eq);
            float* tmp;
            tmp = hi; hi = ho; ho = tmp;
            tmp = ci; ci = co; co = tmp;
        }
        float *h2fin = hi, *c2fin = ci;

        // blend + global L2 normalize -> decoder initial states
        blend_norm_kernel<<<1, 1024, 0, stream>>>(h1fin, h2fin, HsA + (size_t)l * BH);
        blend_norm_kernel<<<1, 1024, 0, stream>>>(c1fin, c2fin, CsA + (size_t)l * BH);

        float* tmp;
        tmp = xin1; xin1 = xout1; xout1 = tmp;
        tmp = xin2; xin2 = xout2; xout2 = tmp;
    }

    // ---- decoder ----
    float* HsP[2] = { HsA, HsB };
    float* CsP[2] = { CsA, CsB };
    for (int s = 0; s < SEQ; ++s) {
        const int cur = s & 1, nxt = cur ^ 1;
        dec_embed_kernel<<<Bq, blk256, 0, stream>>>(
            s == 0 ? (const int*)nullptr : tokbuf, /*start_token=*/1, emb1, xdec);
        for (int l = 0; l < Lq; ++l) {
            const float* xl = (l == 0) ? xdec : HsP[nxt] + (size_t)(l - 1) * BH;
            lstm_step_kernel<<<lstmGrid, blk256, 0, stream>>>(
                HsP[cur] + (size_t)l * BH, CsP[cur] + (size_t)l * BH,
                dWhP + l * WLSZ,
                xl, dWxP + l * WLSZ,
                nullptr, 0,
                dbias + (size_t)l * G4H,
                HsP[nxt] + (size_t)l * BH, CsP[nxt] + (size_t)l * BH,
                nullptr, 0);
        }
        // logits straight into d_out: [B, SEQ, V] slice for step s (row stride SEQ*V)
        float* lg = out + (size_t)Bq * SEQ + (size_t)s * Vq;
        wmma_gemm_kernel<<<logitsGrid, blk256, 0, stream>>>(
            HsP[nxt] + (size_t)(Lq - 1) * BH, Hq, WdP, bd, lg, (long long)SEQ * Vq, Hq);
        if (s == 0)
            argmax_kernel<<<1, 1024, 0, stream>>>(lg, (long long)SEQ * Vq, tokbuf, out, s);
        else
            topk_sample_kernel<<<1, 1024, 0, stream>>>(lg, (long long)SEQ * Vq, tokbuf, out, s);
    }
}
